// DSRRNN_MULTI_50723563766116
// MI455X (gfx1250) — matrix operand order repaired and checked by reading, not compiled, not run on hardware
//
#include <hip/hip_runtime.h>
#include <hip/hip_bf16.h>

typedef _Float16 v16h __attribute__((ext_vector_type(16)));
typedef _Float16 v8h  __attribute__((ext_vector_type(8)));
typedef float    v8f  __attribute__((ext_vector_type(8)));
typedef float    v4f  __attribute__((ext_vector_type(4)));

__device__ __forceinline__ float fast_tanh(float x) {
#if __has_builtin(__builtin_amdgcn_tanhf)
    return __builtin_amdgcn_tanhf(x);
#else
    float e = __expf(2.0f * x);
    return 1.0f - 2.0f / (e + 1.0f);
#endif
}

// ---------------------------------------------------------------------------
// Kernel 0: convert weights to f16 with padding, fold biases.
//   whh16  [256][256]  : whh16[n][k] = W_hh[n][k]          (B-matrix source)
//   wih16  [256][64]   : wih16[n][k] = k<40 ? W_ih[n][k]:0
//   wproj16[32][256]   : wproj16[n][k] = n<20 ? W_proj[n][k]:0
//   bsum   [256]       : b_ih + b_hh
//   bprojp [32]        : padded b_proj
// ---------------------------------------------------------------------------
__global__ void rnn_prep_kernel(const float* __restrict__ Wih,
                                const float* __restrict__ Whh,
                                const float* __restrict__ bih,
                                const float* __restrict__ bhh,
                                const float* __restrict__ Wproj,
                                const float* __restrict__ bproj,
                                _Float16* __restrict__ whh16,
                                _Float16* __restrict__ wih16,
                                _Float16* __restrict__ wproj16,
                                float* __restrict__ bsum,
                                float* __restrict__ bprojp) {
    int tid = blockIdx.x * blockDim.x + threadIdx.x;
    if (tid < 256 * 256) whh16[tid] = (_Float16)Whh[tid];
    if (tid < 256 * 64) {
        int n = tid >> 6, k = tid & 63;
        wih16[tid] = (k < 40) ? (_Float16)Wih[n * 40 + k] : (_Float16)0.0f;
    }
    if (tid < 32 * 256) {
        int n = tid >> 8, k = tid & 255;
        wproj16[tid] = (n < 20) ? (_Float16)Wproj[n * 256 + k] : (_Float16)0.0f;
    }
    if (tid < 256) bsum[tid] = bih[tid] + bhh[tid];
    if (tid < 32)  bprojp[tid] = (tid < 20) ? bproj[tid] : 0.0f;
}

// ---------------------------------------------------------------------------
// Kernel 1: fused RNN step + projection + softmax.
// 128 threads (4 wave32) per block; 64 rows per block; each wave owns 16 rows.
// Two column tiles processed concurrently (dual accumulators) so two
// B-fragment load pairs are in flight while WMMAs retire on the other chain.
// ---------------------------------------------------------------------------
__global__ __launch_bounds__(128)
void rnn_step_kernel(const float* __restrict__ X,
                     const float* __restrict__ H,
                     const _Float16* __restrict__ whh16,
                     const _Float16* __restrict__ wih16,
                     const _Float16* __restrict__ wproj16,
                     const float* __restrict__ bsum,
                     const float* __restrict__ bprojp,
                     float* __restrict__ probs,
                     float* __restrict__ hnew_out) {
    __shared__ _Float16 lds_h[64 * 256];     // h_new f16, row-major, 32 KB
    __shared__ float    lds_logit[64 * 32];  // logits padded to 32, 8 KB

    const int tid  = threadIdx.x;
    const int lane = tid & 31;
    const int wave = tid >> 5;
    const int nIdx = lane & 15;       // column within tile / A-row within tile
    const int hlf  = lane >> 4;       // wave half (0: lanes 0-15, 1: 16-31)
    const int row0 = blockIdx.x * 64 + wave * 16;
    const int mrow = row0 + nIdx;     // this lane's A-matrix row

    const float* Hrow = H + (size_t)mrow * 256;
    const float* Xrow = X + (size_t)mrow * 40;

    // ---- A fragments from h (K=256 -> 8 steps of 32), f32->f16 on the fly.
    // 16-bit A layout used here (contiguous, same as B): element e <-> k_local = hlf*16 + e
    v16h aH[8];
#pragma unroll
    for (int ks = 0; ks < 8; ++ks) {
        const int base = ks * 32 + hlf * 16;
        const v4f f0 = *(const v4f*)(Hrow + base);
        const v4f f1 = *(const v4f*)(Hrow + base + 4);
        const v4f f2 = *(const v4f*)(Hrow + base + 8);
        const v4f f3 = *(const v4f*)(Hrow + base + 12);
#pragma unroll
        for (int e = 0; e < 4; ++e) {
            aH[ks][e]      = (_Float16)f0[e];
            aH[ks][e + 4]  = (_Float16)f1[e];
            aH[ks][e + 8]  = (_Float16)f2[e];
            aH[ks][e + 12] = (_Float16)f3[e];
        }
    }

    // ---- A fragments from x (K=40 padded to 64 -> 2 steps of 32)
    v16h aX[2];
#pragma unroll
    for (int ks = 0; ks < 2; ++ks) {
#pragma unroll
        for (int e = 0; e < 16; ++e) {
            const int k = ks * 32 + hlf * 16 + e;
            aX[ks][e] = (k < 40) ? (_Float16)Xrow[k] : (_Float16)0.0f;
        }
    }

    // ---- 16 column tiles of 16 (pairs of 2 in flight) -> 256-wide hidden
    for (int ct = 0; ct < 16; ct += 2) {
        const int n0 = ct * 16 + nIdx;
        const int n1 = n0 + 16;
        const float bs0 = bsum[n0];
        const float bs1 = bsum[n1];
        v8f acc0, acc1;
#pragma unroll
        for (int r = 0; r < 8; ++r) { acc0[r] = bs0; acc1[r] = bs1; }

        // h @ W_hh^T : B[k][n] = W_hh[n][k]; 32B contiguous per lane per step
        const _Float16* w0 = whh16 + n0 * 256 + hlf * 16;
        const _Float16* w1 = whh16 + n1 * 256 + hlf * 16;
#pragma unroll
        for (int ks = 0; ks < 8; ++ks) {
            v16h b0 = *(const v16h*)(w0 + ks * 32);
            v16h b1 = *(const v16h*)(w1 + ks * 32);
            acc0 = __builtin_amdgcn_wmma_f32_16x16x32_f16(false, aH[ks], false, b0,
                                                          (short)0, acc0, false, false);
            acc1 = __builtin_amdgcn_wmma_f32_16x16x32_f16(false, aH[ks], false, b1,
                                                          (short)0, acc1, false, false);
        }
        // x @ W_ih^T (padded K)
        const _Float16* wi0 = wih16 + n0 * 64 + hlf * 16;
        const _Float16* wi1 = wih16 + n1 * 64 + hlf * 16;
#pragma unroll
        for (int ks = 0; ks < 2; ++ks) {
            v16h b0 = *(const v16h*)(wi0 + ks * 32);
            v16h b1 = *(const v16h*)(wi1 + ks * 32);
            acc0 = __builtin_amdgcn_wmma_f32_16x16x32_f16(false, aX[ks], false, b0,
                                                          (short)0, acc0, false, false);
            acc1 = __builtin_amdgcn_wmma_f32_16x16x32_f16(false, aX[ks], false, b1,
                                                          (short)0, acc1, false, false);
        }

        // tanh, write h_new (D layout: VGPR r <-> m = hlf*8 + r)
#pragma unroll
        for (int r = 0; r < 8; ++r) {
            const float hv0 = fast_tanh(acc0[r]);
            const float hv1 = fast_tanh(acc1[r]);
            const int grow = row0 + hlf * 8 + r;
            hnew_out[(size_t)grow * 256 + n0] = hv0;
            hnew_out[(size_t)grow * 256 + n1] = hv1;
            lds_h[(wave * 16 + hlf * 8 + r) * 256 + n0] = (_Float16)hv0;
            lds_h[(wave * 16 + hlf * 8 + r) * 256 + n1] = (_Float16)hv1;
        }
    }
    __syncthreads();

    // ---- projection: logits = h_new @ W_proj^T + b (20 padded to 32 cols)
    v16h aP[8];
    const _Float16* hrow_l = lds_h + (wave * 16 + nIdx) * 256;
#pragma unroll
    for (int ks = 0; ks < 8; ++ks) {
        const int base = ks * 32 + hlf * 16;
        v8h p0 = *(const v8h*)(hrow_l + base);
        v8h p1 = *(const v8h*)(hrow_l + base + 8);
#pragma unroll
        for (int e = 0; e < 8; ++e) {
            aP[ks][e]     = p0[e];
            aP[ks][e + 8] = p1[e];
        }
    }
    {
        const int n0 = nIdx;
        const int n1 = nIdx + 16;
        const float bp0 = bprojp[n0];
        const float bp1 = bprojp[n1];
        v8f acc0, acc1;
#pragma unroll
        for (int r = 0; r < 8; ++r) { acc0[r] = bp0; acc1[r] = bp1; }
        const _Float16* w0 = wproj16 + n0 * 256 + hlf * 16;
        const _Float16* w1 = wproj16 + n1 * 256 + hlf * 16;
#pragma unroll
        for (int ks = 0; ks < 8; ++ks) {
            v16h b0 = *(const v16h*)(w0 + ks * 32);
            v16h b1 = *(const v16h*)(w1 + ks * 32);
            acc0 = __builtin_amdgcn_wmma_f32_16x16x32_f16(false, aP[ks], false, b0,
                                                          (short)0, acc0, false, false);
            acc1 = __builtin_amdgcn_wmma_f32_16x16x32_f16(false, aP[ks], false, b1,
                                                          (short)0, acc1, false, false);
        }
#pragma unroll
        for (int r = 0; r < 8; ++r) {
            lds_logit[(wave * 16 + hlf * 8 + r) * 32 + n0] = acc0[r];
            lds_logit[(wave * 16 + hlf * 8 + r) * 32 + n1] = acc1[r];
        }
    }
    __syncthreads();

    // ---- softmax over the 20 valid logits, one row per thread (64 rows)
    if (tid < 64) {
        const float* lrow = lds_logit + tid * 32;
        float mx = lrow[0];
#pragma unroll
        for (int j = 1; j < 20; ++j) mx = fmaxf(mx, lrow[j]);
        float ev[20];
        float s = 0.0f;
#pragma unroll
        for (int j = 0; j < 20; ++j) { ev[j] = __expf(lrow[j] - mx); s += ev[j]; }
        const float inv = 1.0f / s;
        const int grow = blockIdx.x * 64 + tid;
        float* prow = probs + (size_t)grow * 20;
#pragma unroll
        for (int j = 0; j < 20; ++j) prow[j] = ev[j] * inv;
    }
}

extern "C" void kernel_launch(void* const* d_in, const int* in_sizes, int n_in,
                              void* d_out, int out_size, void* d_ws, size_t ws_size,
                              hipStream_t stream) {
    const float* x     = (const float*)d_in[0];
    const float* h     = (const float*)d_in[1];
    const float* Wih   = (const float*)d_in[2];
    const float* Whh   = (const float*)d_in[3];
    const float* bih   = (const float*)d_in[4];
    const float* bhh   = (const float*)d_in[5];
    const float* Wproj = (const float*)d_in[6];
    const float* bproj = (const float*)d_in[7];

    const int n = in_sizes[1] / 256;   // 262144

    char* ws = (char*)d_ws;
    _Float16* whh16   = (_Float16*)(ws);            // 131072 B
    _Float16* wih16   = (_Float16*)(ws + 131072);   //  32768 B
    _Float16* wproj16 = (_Float16*)(ws + 163840);   //  16384 B
    float*    bsum    = (float*)(ws + 180224);      //   1024 B
    float*    bprojp  = (float*)(ws + 181248);      //    128 B

    float* probs = (float*)d_out;
    float* hnew  = probs + (size_t)n * 20;

    rnn_prep_kernel<<<256, 256, 0, stream>>>(Wih, Whh, bih, bhh, Wproj, bproj,
                                             whh16, wih16, wproj16, bsum, bprojp);
    rnn_step_kernel<<<n / 64, 128, 0, stream>>>(x, h, whh16, wih16, wproj16,
                                                bsum, bprojp, probs, hnew);
}
